// STELLALayer_55808805044757
// MI455X (gfx1250) — matrix operand order repaired and checked by reading, not compiled, not run on hardware
//
#include <hip/hip_runtime.h>
#include <hip/hip_bf16.h>

// ---------------- problem constants ----------------
constexpr int Bc = 4, Sc = 1024, Hc = 768;
constexpr int NHc = 12, HDc = 64;
constexpr int Ec = 8, Fc = 384, SFc = 768;
constexpr int Tc = Bc * Sc;                 // 4096 tokens

typedef _Float16 h16 __attribute__((ext_vector_type(16)));
typedef _Float16 h8  __attribute__((ext_vector_type(8)));
typedef float    f8  __attribute__((ext_vector_type(8)));

__device__ __forceinline__ h16 cat8(h8 lo, h8 hi) {
    return __builtin_shufflevector(lo, hi, 0,1,2,3,4,5,6,7,8,9,10,11,12,13,14,15);
}

// A-fragment (16x32 f16): lane m=lane%16 row, half=lane/16.
// K values per lane: K = half*16 + e  (one 32B run)
__device__ __forceinline__ h16 frag_a(const _Float16* rowbase, int ldp) {
    int lane = threadIdx.x & 31;
    int m = lane & 15, half = lane >> 4;
    const _Float16* p = rowbase + m * ldp;
    h8 lo = *(const h8*)(p + half * 16);
    h8 hi = *(const h8*)(p + 8 + half * 16);
    return cat8(lo, hi);
}

// B-fragment (32x16 f16): lane n=lane%16 col, K = half*16 + e  (one 32B run)
__device__ __forceinline__ h16 frag_b(const _Float16* rowbase, int ldp) {
    int lane = threadIdx.x & 31;
    int n = lane & 15, half = lane >> 4;
    const _Float16* p = rowbase + n * ldp + half * 16;
    return cat8(*(const h8*)(p), *(const h8*)(p + 8));
}

// ---- CDNA5 async memory->LDS path (ASYNCcnt tracked, no VGPR data movement) ----
// Per-lane semantics: copies 16 bytes from the global address to the LDS address.
__device__ __forceinline__ void async_load_b128(const _Float16* gptr, _Float16* lptr) {
    unsigned lds = (unsigned)(size_t)(void*)lptr;            // LDS addr = generic[31:0]
    unsigned long long ga = (unsigned long long)(size_t)(const void*)gptr;
    asm volatile("global_load_async_to_lds_b128 %0, %1, off"
                 :: "v"(lds), "v"(ga) : "memory");
}
__device__ __forceinline__ void wait_async0() {
    asm volatile("s_wait_asynccnt 0x0" ::: "memory");
}
// Drain all but the newest 4 async ops (loads retire in order -> previous tile done)
__device__ __forceinline__ void wait_async4() {
    asm volatile("s_wait_asynccnt 0x4" ::: "memory");
}

// =====================================================================
// Tiled WMMA GEMM:  C[M,N] = A[M,Kd] * W[N,Kd]^T
//   MODE 0 : outH  = f16 result
//   MODE 1 : outF += acc
//   MODE 2 : outF += rowscale[row*rsStride] * acc
// BM=BN=128, BK=32, 256 threads = 8 waves as 2(M) x 4(N); wave tile 64x32.
// Double-buffered LDS tiles fed by global_load_async_to_lds_b128; the
// transfer of tile i+1 overlaps the 8 WMMAs of tile i (s_wait_asynccnt 4).
// =====================================================================
template<int MODE>
__global__ __launch_bounds__(256) void gemm_xwT_wmma(
        const _Float16* __restrict__ A, const _Float16* __restrict__ W,
        _Float16* __restrict__ outH, float* __restrict__ outF,
        const float* __restrict__ rowscale, int rsStride,
        int M, int N, int Kd)
{
    constexpr int LDP = 40;
    __shared__ __align__(16) _Float16 As[2][128 * LDP];
    __shared__ __align__(16) _Float16 Bs[2][128 * LDP];

    const int tid = threadIdx.x;
    const int m0 = blockIdx.y * 128, n0 = blockIdx.x * 128;
    const int wid = tid >> 5, lane = tid & 31;
    const int wm = wid & 1, wn = wid >> 1;

    f8 acc[4][2];
    const f8 z8 = {0.f,0.f,0.f,0.f,0.f,0.f,0.f,0.f};
#pragma unroll
    for (int i = 0; i < 4; ++i)
#pragma unroll
        for (int j = 0; j < 2; ++j) acc[i][j] = z8;

    // per-thread tile-chunk coordinates (2 x 16B chunks each for A and B)
    const int r0c = (tid + 0)   >> 2, c0c = ((tid + 0)   & 3) * 8;
    const int r1c = (tid + 256) >> 2, c1c = ((tid + 256) & 3) * 8;

    auto issue_tile = [&](int buf, int kb) {
        async_load_b128(&A[(size_t)(m0 + r0c) * Kd + kb + c0c], &As[buf][r0c * LDP + c0c]);
        async_load_b128(&A[(size_t)(m0 + r1c) * Kd + kb + c1c], &As[buf][r1c * LDP + c1c]);
        async_load_b128(&W[(size_t)(n0 + r0c) * Kd + kb + c0c], &Bs[buf][r0c * LDP + c0c]);
        async_load_b128(&W[(size_t)(n0 + r1c) * Kd + kb + c1c], &Bs[buf][r1c * LDP + c1c]);
    };

    issue_tile(0, 0);                                   // prologue: tile 0 in flight
    int ib = 0;
    for (int kb = 0; kb < Kd; kb += 32, ib ^= 1) {
        const bool has_next = (kb + 32) < Kd;
        __syncthreads();          // all waves done computing from buf ib^1 (iter i-1)
        if (has_next) issue_tile(ib ^ 1, kb + 32);      // overlaps this tile's math
        if (has_next) wait_async4(); else wait_async0();// current tile's 4 ops done
        __syncthreads();          // ... across all waves

        h16 bfr[2];
#pragma unroll
        for (int nt = 0; nt < 2; ++nt)
            bfr[nt] = frag_b(&Bs[ib][(wn * 32 + nt * 16) * LDP], LDP);
#pragma unroll
        for (int mt = 0; mt < 4; ++mt) {
            h16 afr = frag_a(&As[ib][(wm * 64 + mt * 16) * LDP], LDP);
#pragma unroll
            for (int nt = 0; nt < 2; ++nt)
                acc[mt][nt] = __builtin_amdgcn_wmma_f32_16x16x32_f16(
                    false, afr, false, bfr[nt], (short)0, acc[mt][nt], false, false);
        }
    }

    // D layout: VGPR r -> row (half*8 + r), col = lane%16
    const int half = lane >> 4, nin = lane & 15;
#pragma unroll
    for (int mt = 0; mt < 4; ++mt) {
        const int rowb = m0 + wm * 64 + mt * 16 + half * 8;
#pragma unroll
        for (int nt = 0; nt < 2; ++nt) {
            const int col = n0 + wn * 32 + nt * 16 + nin;
#pragma unroll
            for (int r = 0; r < 8; ++r) {
                const size_t row = (size_t)(rowb + r);
                const float v = acc[mt][nt][r];
                if constexpr (MODE == 0) {
                    outH[row * N + col] = (_Float16)v;
                } else if constexpr (MODE == 1) {
                    outF[row * N + col] += v;
                } else {
                    outF[row * N + col] += rowscale[row * rsStride] * v;
                }
            }
        }
    }
}

// =====================================================================
// Flash-attention with WMMA. Block = 128 threads (4 waves), each wave owns
// 16 query rows of one (batch, head). K/V streamed in 32-key tiles via LDS;
// the K tile rides the async-to-LDS path, V is transposed through registers.
// =====================================================================
__global__ __launch_bounds__(128) void attn_wmma(
        const _Float16* __restrict__ Q, const _Float16* __restrict__ Kx,
        const _Float16* __restrict__ Vx, _Float16* __restrict__ ctxOut)
{
    __shared__ __align__(16) _Float16 Ks[32 * 72];      // [key][hd]
    __shared__ __align__(16) _Float16 Vt[64 * 40];      // [hd][key] (transposed)
    __shared__ __align__(16) _Float16 Ps[4 * 16 * 40];  // per-wave P tile

    const int tid = threadIdx.x, wid = tid >> 5, lane = tid & 31;
    const int half = lane >> 4, lan = lane & 15;
    const int bb = blockIdx.z, hh = blockIdx.y;
    const int q0 = blockIdx.x * 64 + wid * 16;

    // preload q fragments for this wave's 16 rows (HD=64 -> two K-steps)
    const _Float16* qp = Q + ((size_t)(bb * Sc + q0 + lan)) * Hc + hh * HDc;
    h16 aq0 = cat8(*(const h8*)(qp + half * 16),      *(const h8*)(qp + 8 + half * 16));
    h16 aq1 = cat8(*(const h8*)(qp + 32 + half * 16), *(const h8*)(qp + 40 + half * 16));

    float mrow[8], lrow[8];
    f8 cacc[4];
    const f8 z8 = {0.f,0.f,0.f,0.f,0.f,0.f,0.f,0.f};
#pragma unroll
    for (int r = 0; r < 8; ++r) { mrow[r] = -1e30f; lrow[r] = 0.f; }
#pragma unroll
    for (int c = 0; c < 4; ++c) cacc[c] = z8;

    _Float16* Pw = &Ps[wid * 16 * 40];

    for (int kt = 0; kt < Sc; kt += 32) {
        __syncthreads();
#pragma unroll
        for (int i = 0; i < 2; ++i) {
            int idx = tid + i * 128;            // 256 16B vectors
            int r = idx >> 3, c = (idx & 7) * 8;
            size_t gro = ((size_t)(bb * Sc + kt + r)) * Hc + hh * HDc + c;
            async_load_b128(&Kx[gro], &Ks[r * 72 + c]);   // K: async, no transpose
            h8 vv = *(const h8*)&Vx[gro];                 // V: register transpose
#pragma unroll
            for (int j = 0; j < 8; ++j) Vt[(c + j) * 40 + r] = vv[j];
        }
        wait_async0();
        __syncthreads();

        // scores: 16 q-rows x 32 keys  (2 N-tiles x 2 K-steps)
        f8 sacc[2] = {z8, z8};
#pragma unroll
        for (int nt = 0; nt < 2; ++nt)
#pragma unroll
            for (int ks = 0; ks < 2; ++ks) {
                const _Float16* bp = &Ks[(nt * 16 + lan) * 72 + ks * 32 + half * 16];
                h16 bf = cat8(*(const h8*)bp, *(const h8*)(bp + 8));
                sacc[nt] = __builtin_amdgcn_wmma_f32_16x16x32_f16(
                    false, ks ? aq1 : aq0, false, bf, (short)0, sacc[nt], false, false);
            }

        // online softmax (row stats replicated across the 16 lanes of a half-group)
#pragma unroll
        for (int r = 0; r < 8; ++r) {
            float s0 = sacc[0][r] * 0.125f, s1 = sacc[1][r] * 0.125f;
            float mx = fmaxf(s0, s1);
#pragma unroll
            for (int o = 1; o < 16; o <<= 1) mx = fmaxf(mx, __shfl_xor(mx, o, 32));
            float mnew  = fmaxf(mrow[r], mx);
            float alpha = __expf(mrow[r] - mnew);
            float p0 = __expf(s0 - mnew), p1 = __expf(s1 - mnew);
            float sm = p0 + p1;
#pragma unroll
            for (int o = 1; o < 16; o <<= 1) sm += __shfl_xor(sm, o, 32);
            lrow[r] = lrow[r] * alpha + sm;
            mrow[r] = mnew;
#pragma unroll
            for (int c = 0; c < 4; ++c) cacc[c][r] *= alpha;   // same row mapping as scores
            Pw[(half * 8 + r) * 40 + lan]      = (_Float16)p0;
            Pw[(half * 8 + r) * 40 + 16 + lan] = (_Float16)p1;
        }
        __syncthreads();

        // ctx += P(16x32) @ V(32x64)
        h16 ap = frag_a(Pw, 40);
#pragma unroll
        for (int c = 0; c < 4; ++c) {
            const _Float16* bp = &Vt[(c * 16 + lan) * 40 + half * 16];
            h16 bf = cat8(*(const h8*)bp, *(const h8*)(bp + 8));
            cacc[c] = __builtin_amdgcn_wmma_f32_16x16x32_f16(
                false, ap, false, bf, (short)0, cacc[c], false, false);
        }
    }

#pragma unroll
    for (int r = 0; r < 8; ++r) {
        float inv = 1.0f / lrow[r];
        size_t row = (size_t)(bb * Sc + q0 + half * 8 + r);
#pragma unroll
        for (int c = 0; c < 4; ++c)
            ctxOut[row * Hc + hh * HDc + c * 16 + lan] = (_Float16)(cacc[c][r] * inv);
    }
}

// ---------------- elementwise / reduction kernels ----------------
__global__ __launch_bounds__(256) void cast_f32_f16(const float* __restrict__ in,
                                                    _Float16* __restrict__ out, size_t n)
{
    size_t i = (size_t)blockIdx.x * 256 + threadIdx.x;
    if (i < n) out[i] = (_Float16)in[i];
}

__global__ __launch_bounds__(256) void rmsnorm_f16(const float* __restrict__ x,
                                                   const float* __restrict__ w,
                                                   _Float16* __restrict__ out, int Hd)
{
    __shared__ float red[256];
    const int t = blockIdx.x;
    const float* xr = x + (size_t)t * Hd;
    float ss = 0.f;
    for (int i = threadIdx.x; i < Hd; i += 256) { float v = xr[i]; ss += v * v; }
    red[threadIdx.x] = ss;
    __syncthreads();
    for (int s = 128; s > 0; s >>= 1) {
        if (threadIdx.x < s) red[threadIdx.x] += red[threadIdx.x + s];
        __syncthreads();
    }
    float inv = rsqrtf(red[0] / (float)Hd + 1e-6f);
    for (int i = threadIdx.x; i < Hd; i += 256)
        out[(size_t)t * Hd + i] = (_Float16)(xr[i] * inv * w[i]);
}

__global__ __launch_bounds__(256) void swiglu_f16(const _Float16* __restrict__ g,
                                                  const _Float16* __restrict__ u,
                                                  _Float16* __restrict__ a, size_t n)
{
    size_t i = (size_t)blockIdx.x * 256 + threadIdx.x;
    if (i < n) {
        float gv = (float)g[i];
        a[i] = (_Float16)((gv / (1.f + __expf(-gv))) * (float)u[i]);
    }
}

__global__ __launch_bounds__(256) void gate_topk(const _Float16* __restrict__ xn,
                                                 const float* __restrict__ gw,
                                                 float* __restrict__ combine, int Hd)
{
    int t = blockIdx.x * 256 + threadIdx.x;
    if (t >= Tc) return;
    float lg[Ec];
    const _Float16* xr = xn + (size_t)t * Hd;
    for (int e = 0; e < Ec; ++e) {
        const float* ge = gw + (size_t)e * Hd;
        float s = 0.f;
        for (int i = 0; i < Hd; ++i) s += (float)xr[i] * ge[i];
        lg[e] = s;
    }
    float mx = lg[0];
    for (int e = 1; e < Ec; ++e) mx = fmaxf(mx, lg[e]);
    float sum = 0.f;
    for (int e = 0; e < Ec; ++e) { lg[e] = __expf(lg[e] - mx); sum += lg[e]; }
    float invs = 1.f / sum;
    for (int e = 0; e < Ec; ++e) lg[e] *= invs;      // softmax probs
    int i1 = 0;
    for (int e = 1; e < Ec; ++e) if (lg[e] > lg[i1]) i1 = e;
    int i2 = (i1 == 0) ? 1 : 0;
    for (int e = 0; e < Ec; ++e) if (e != i1 && lg[e] > lg[i2]) i2 = e;
    float p1 = lg[i1], p2 = lg[i2];
    float wn = 1.f / (p1 + p2 + 1e-20f);
    for (int e = 0; e < Ec; ++e) combine[(size_t)t * Ec + e] = 0.f;
    combine[(size_t)t * Ec + i1] = p1 * wn;
    combine[(size_t)t * Ec + i2] = p2 * wn;
}

// =====================================================================
extern "C" void kernel_launch(void* const* d_in, const int* in_sizes, int n_in,
                              void* d_out, int out_size, void* d_ws, size_t ws_size,
                              hipStream_t stream)
{
    (void)in_sizes; (void)n_in; (void)out_size; (void)ws_size;
    const float* hidden = (const float*)d_in[0];
    const float* n1w    = (const float*)d_in[1];
    const float* Wq     = (const float*)d_in[2];
    const float* Wk     = (const float*)d_in[3];
    const float* Wv     = (const float*)d_in[4];
    const float* Wo     = (const float*)d_in[5];
    const float* n2w    = (const float*)d_in[6];
    const float* gatew  = (const float*)d_in[7];
    const float* Weg    = (const float*)d_in[8];
    const float* Weu    = (const float*)d_in[9];
    const float* Wed    = (const float*)d_in[10];
    const float* Wsg    = (const float*)d_in[11];
    const float* Wsu    = (const float*)d_in[12];
    const float* Wsd    = (const float*)d_in[13];
    float* out = (float*)d_out;

    const size_t nHH = (size_t)Hc * Hc, nEFH = (size_t)Ec * Fc * Hc;
    const size_t nSFH = (size_t)SFc * Hc, nTH = (size_t)Tc * Hc, nTF = (size_t)Tc * Fc;

    char* base = (char*)d_ws; size_t off = 0;
    auto alloc = [&](size_t bytes) -> void* {
        void* p = base + off; off = (off + bytes + 255) & ~(size_t)255; return p;
    };
    _Float16* hWq  = (_Float16*)alloc(nHH * 2);
    _Float16* hWk  = (_Float16*)alloc(nHH * 2);
    _Float16* hWv  = (_Float16*)alloc(nHH * 2);
    _Float16* hWo  = (_Float16*)alloc(nHH * 2);
    _Float16* hWeg = (_Float16*)alloc(nEFH * 2);
    _Float16* hWeu = (_Float16*)alloc(nEFH * 2);
    _Float16* hWed = (_Float16*)alloc(nEFH * 2);
    _Float16* hWsg = (_Float16*)alloc(nSFH * 2);
    _Float16* hWsu = (_Float16*)alloc(nSFH * 2);
    _Float16* hWsd = (_Float16*)alloc(nSFH * 2);
    _Float16* xn1  = (_Float16*)alloc(nTH * 2);
    _Float16* qb   = (_Float16*)alloc(nTH * 2);
    _Float16* kb   = (_Float16*)alloc(nTH * 2);
    _Float16* vb   = (_Float16*)alloc(nTH * 2);
    _Float16* ctx  = (_Float16*)alloc(nTH * 2);
    _Float16* xn2  = (_Float16*)alloc(nTH * 2);
    _Float16* gsh  = (_Float16*)alloc((size_t)Tc * SFc * 2);
    _Float16* ush  = (_Float16*)alloc((size_t)Tc * SFc * 2);
    _Float16* ash  = (_Float16*)alloc((size_t)Tc * SFc * 2);
    _Float16* grt  = (_Float16*)alloc(nTF * 2);
    _Float16* urt  = (_Float16*)alloc(nTF * 2);
    _Float16* art  = (_Float16*)alloc(nTF * 2);
    float* x1      = (float*)alloc(nTH * 4);
    float* combine = (float*)alloc((size_t)Tc * Ec * 4);

    auto cast = [&](const float* s, _Float16* d, size_t n) {
        cast_f32_f16<<<(unsigned)((n + 255) / 256), 256, 0, stream>>>(s, d, n);
    };
    auto gemmH = [&](const _Float16* A, const _Float16* W_, _Float16* O,
                     int M, int N, int Kd) {
        gemm_xwT_wmma<0><<<dim3((unsigned)(N / 128), (unsigned)(M / 128)), 256, 0, stream>>>(
            A, W_, O, nullptr, nullptr, 0, M, N, Kd);
    };
    auto gemmAcc = [&](const _Float16* A, const _Float16* W_, float* O,
                       int M, int N, int Kd) {
        gemm_xwT_wmma<1><<<dim3((unsigned)(N / 128), (unsigned)(M / 128)), 256, 0, stream>>>(
            A, W_, nullptr, O, nullptr, 0, M, N, Kd);
    };
    auto gemmAccScaled = [&](const _Float16* A, const _Float16* W_, float* O,
                             const float* rs, int rsS, int M, int N, int Kd) {
        gemm_xwT_wmma<2><<<dim3((unsigned)(N / 128), (unsigned)(M / 128)), 256, 0, stream>>>(
            A, W_, nullptr, O, rs, rsS, M, N, Kd);
    };

    // weight conversion
    cast(Wq, hWq, nHH);  cast(Wk, hWk, nHH);  cast(Wv, hWv, nHH);  cast(Wo, hWo, nHH);
    cast(Weg, hWeg, nEFH); cast(Weu, hWeu, nEFH); cast(Wed, hWed, nEFH);
    cast(Wsg, hWsg, nSFH); cast(Wsu, hWsu, nSFH); cast(Wsd, hWsd, nSFH);

    // ---- attention sublayer ----
    rmsnorm_f16<<<Tc, 256, 0, stream>>>(hidden, n1w, xn1, Hc);
    gemmH(xn1, hWq, qb, Tc, Hc, Hc);
    gemmH(xn1, hWk, kb, Tc, Hc, Hc);
    gemmH(xn1, hWv, vb, Tc, Hc, Hc);
    attn_wmma<<<dim3(Sc / 64, NHc, Bc), 128, 0, stream>>>(qb, kb, vb, ctx);
    hipMemcpyAsync(x1, hidden, nTH * 4, hipMemcpyDeviceToDevice, stream);
    gemmAcc(ctx, hWo, x1, Tc, Hc, Hc);               // x1 = residual + ctx @ Wo^T

    // ---- MoE sublayer ----
    rmsnorm_f16<<<Tc, 256, 0, stream>>>(x1, n2w, xn2, Hc);
    gate_topk<<<Tc / 256, 256, 0, stream>>>(xn2, gatew, combine, Hc);

    // shared expert
    gemmH(xn2, hWsg, gsh, Tc, SFc, Hc);
    gemmH(xn2, hWsu, ush, Tc, SFc, Hc);
    swiglu_f16<<<(unsigned)(((size_t)Tc * SFc) / 256), 256, 0, stream>>>(
        gsh, ush, ash, (size_t)Tc * SFc);
    hipMemcpyAsync(out, x1, nTH * 4, hipMemcpyDeviceToDevice, stream);  // out = res2
    gemmAcc(ash, hWsd, out, Tc, Hc, SFc);                               // += shared

    // routed experts (dense over all tokens, per-row combine scaling)
    for (int e = 0; e < Ec; ++e) {
        gemmH(xn2, hWeg + (size_t)e * Fc * Hc, grt, Tc, Fc, Hc);
        gemmH(xn2, hWeu + (size_t)e * Fc * Hc, urt, Tc, Fc, Hc);
        swiglu_f16<<<(unsigned)(nTF / 256), 256, 0, stream>>>(grt, urt, art, nTF);
        gemmAccScaled(art, hWed + (size_t)e * Hc * Fc, out, combine + e, Ec, Tc, Hc, Fc);
    }
}
